// GNN_old_45904610459951
// MI455X (gfx1250) — compile-verified, matrix operand order checked by reading, not run on hardware
//
#include <hip/hip_runtime.h>
#include <hip/hip_bf16.h>

typedef __attribute__((ext_vector_type(16))) _Float16 v16h;
typedef __attribute__((ext_vector_type(8)))  _Float16 v8h;
typedef __attribute__((ext_vector_type(8)))  float    v8f;

#define NNODES 100000
#define NEDGES 3200000

// ---------------------------------------------------------------------------
// Weight pre-swizzle: pack W[K][256] into WMMA B-fragment order so that for
// fragment (ntile, ktile), lane l's 16 half elements are 32 contiguous bytes:
//   Wp[ (((ntile*KT + ktile)*32 + lane)*16 + i ] , KT = K/32
// Forward map (lane l, element i): n = ntile*16 + (l&15),
//   k = ktile*32 + (i>>3)*16 + (l>>4)*8 + (i&7)
// Inverse (k, n): lane = ((kin>>3)&1)*16 + (n&15), i = (kin>>4)*8 + (kin&7)
// ---------------------------------------------------------------------------
__device__ __forceinline__ void pack_one(const float* __restrict__ W,
                                         _Float16* __restrict__ Wp,
                                         int k, int n, int K) {
    const int ntile = n >> 4, m = n & 15;
    const int ktile = k >> 5, kin = k & 31;
    const int half = (kin >> 3) & 1;
    const int i    = ((kin >> 4) << 3) | (kin & 7);
    const int lane = (half << 4) | m;
    const int KT   = K >> 5;
    Wp[((size_t)(ntile * KT + ktile) * 32 + lane) * 16 + i] = (_Float16)W[k * 256 + n];
}

__global__ void k_cvt_weights(const float* __restrict__ W1, const float* __restrict__ W2,
                              const float* __restrict__ W3, const float* __restrict__ W4,
                              _Float16* __restrict__ W1p, _Float16* __restrict__ W2p,
                              _Float16* __restrict__ W3p, _Float16* __restrict__ W4h) {
    const int idx = blockIdx.x * blockDim.x + threadIdx.x;
    const int k = idx >> 8, n = idx & 255;           // all W's have row length 256
    if (idx < 128 * 256) pack_one(W1, W1p, k, n, 128);
    if (idx < 256 * 256) { pack_one(W2, W2p, k, n, 256); pack_one(W3, W3p, k, n, 256); }
    if (idx < 256)       W4h[idx] = (_Float16)W4[idx];
}

// ---------------------------------------------------------------------------
// Edge scatter. One wave per edge: lanes 0..30 read x as float4, lane 31
// packs pos[0..2] + z. 4 f32 atomics per lane into sums[src]. sums + x +
// edge_index all fit in 192MB L2 -> random traffic stays at L2 speed.
// ---------------------------------------------------------------------------
__global__ void k_scatter(const float* __restrict__ x, const float* __restrict__ pos,
                          const float* __restrict__ z, const long long* __restrict__ ei,
                          float* __restrict__ sums) {
    int tid  = blockIdx.x * blockDim.x + threadIdx.x;
    int e    = tid >> 5;
    if (e >= NEDGES) return;
    int lane = tid & 31;
    int src  = (int)ei[e];              // edge_index[0][e]
    int dst  = (int)ei[NEDGES + e];     // edge_index[1][e]
    float4 v;
    if (lane < 31) {
        v = *(const float4*)(x + (long long)dst * 124 + lane * 4);
    } else {
        const float* p = pos + (long long)dst * 3;
        v.x = p[0]; v.y = p[1]; v.z = p[2]; v.w = z[dst];
    }
    float* srow = sums + (long long)src * 128 + lane * 4;
    atomicAdd(srow + 0, v.x);
    atomicAdd(srow + 1, v.y);
    atomicAdd(srow + 2, v.z);
    atomicAdd(srow + 3, v.w);
}

// ---------------------------------------------------------------------------
// A-fragment from LDS (16-bit A 16x32 layout): lane holds row m = l&15;
// element i -> K = (i>>3)*16 + (l>>4)*8 + (i&7) => two contiguous 16B chunks.
// ---------------------------------------------------------------------------
__device__ __forceinline__ v16h load_a_frag(const _Float16* Arow, int kbase, int half) {
    v8h lo = *(const v8h*)(Arow + kbase + half * 8);
    v8h hi = *(const v8h*)(Arow + kbase + 16 + half * 8);
    v16h a;
#pragma unroll
    for (int i = 0; i < 8; ++i) { a[i] = lo[i]; a[i + 8] = hi[i]; }
    return a;
}

// One dense layer: [16 x K] (LDS f16) @ [K x 256] (pre-swizzled global f16)
// + bias, ReLU -> LDS f16. 4 waves x 4 N-tiles = all 16 output tiles.
// A-fragments hoisted into registers once; B-fragments are single 32B
// contiguous per-lane loads (2x global_load_b128, fully coalesced per wave).
template <int K>
__device__ __forceinline__ void mlp_layer(const _Float16 (*A)[256],
                                          const _Float16* __restrict__ Wp,
                                          const float* __restrict__ bias,
                                          _Float16 (*Out)[256], int wave, int lane) {
    constexpr int KT = K / 32;
    const int m = lane & 15, half = lane >> 4;

    v16h afrag[KT];
#pragma unroll
    for (int ks = 0; ks < KT; ++ks)
        afrag[ks] = load_a_frag(&A[m][0], ks * 32, half);

#pragma unroll
    for (int t = 0; t < 4; ++t) {
        const int nt = wave * 4 + t;
        const int n  = nt * 16 + m;
        v8f acc = {};
#pragma unroll
        for (int ks = 0; ks < KT; ++ks) {
            const _Float16* src = Wp + ((size_t)(nt * KT + ks) * 32 + lane) * 16;
            v16h b = *(const v16h*)src;   // 32B contiguous -> 2x b128
            acc = __builtin_amdgcn_wmma_f32_16x16x32_f16(
                false, afrag[ks], false, b, (short)0, acc, false, false);
        }
        const float bi = bias[n];
#pragma unroll
        for (int r = 0; r < 8; ++r) {
            float v = acc[r] + bi;
            v = v > 0.f ? v : 0.f;
            Out[r + 8 * half][n] = (_Float16)v;
        }
    }
}

// ---------------------------------------------------------------------------
// Fused per-16-node-tile pipeline: L2-normalize -> 3x WMMA layers (ping-pong
// f16 activations in LDS) -> 256->1 dot -> atomic partial mean.
// ---------------------------------------------------------------------------
__global__ void __launch_bounds__(128)
k_mlp(const float* __restrict__ sums,
      const _Float16* __restrict__ W1p, const float* __restrict__ b1,
      const _Float16* __restrict__ W2p, const float* __restrict__ b2,
      const _Float16* __restrict__ W3p, const float* __restrict__ b3,
      const _Float16* __restrict__ W4h, const float* __restrict__ b4,
      float* __restrict__ accum) {
    __shared__ float    raw[16][128];
    __shared__ float    rn[16];
    __shared__ _Float16 bufA[16][256];
    __shared__ _Float16 bufB[16][256];
    __shared__ float    part[16];

    const int tid = threadIdx.x;
    const long long base = (long long)blockIdx.x * 16 * 128;

    // load 16 node rows (coalesced)
    for (int i = tid; i < 16 * 128; i += 128)
        raw[i >> 7][i & 127] = sums[base + i];
    __syncthreads();

    // per-row L2 norm
    if (tid < 16) {
        float s = 0.f;
        for (int c = 0; c < 128; ++c) { float v = raw[tid][c]; s += v * v; }
        rn[tid] = 1.0f / sqrtf(s);
    }
    __syncthreads();

    // normalize + f16 convert into bufA (K=128 used)
    for (int i = tid; i < 16 * 128; i += 128) {
        int r = i >> 7, c = i & 127;
        bufA[r][c] = (_Float16)(raw[r][c] * rn[r]);
    }
    __syncthreads();

    const int wave = tid >> 5, lane = tid & 31;
    mlp_layer<128>(bufA, W1p, b1, bufB, wave, lane);
    __syncthreads();
    mlp_layer<256>(bufB, W2p, b2, bufA, wave, lane);
    __syncthreads();
    mlp_layer<256>(bufA, W3p, b3, bufB, wave, lane);
    __syncthreads();

    // final 256 -> 1 layer + tile partial sum
    if (tid < 16) {
        float s = 0.f;
        for (int c = 0; c < 256; ++c)
            s += (float)bufB[tid][c] * (float)W4h[c];
        part[tid] = s + b4[0];
    }
    __syncthreads();
    if (tid == 0) {
        float s = 0.f;
        for (int r = 0; r < 16; ++r) s += part[r];
        atomicAdd(accum, s);
    }
}

// ---------------------------------------------------------------------------
__global__ void k_finalize(const float* __restrict__ accum, float* __restrict__ out) {
    if (blockIdx.x == 0 && threadIdx.x == 0)
        out[0] = accum[0] * (1.0f / (float)NNODES);
}

// ---------------------------------------------------------------------------
extern "C" void kernel_launch(void* const* d_in, const int* in_sizes, int n_in,
                              void* d_out, int out_size, void* d_ws, size_t ws_size,
                              hipStream_t stream) {
    (void)in_sizes; (void)n_in; (void)out_size; (void)ws_size;
    const float*      x   = (const float*)d_in[0];
    const float*      pos = (const float*)d_in[1];
    const float*      z   = (const float*)d_in[2];
    const long long*  ei  = (const long long*)d_in[3];
    const float*      W1  = (const float*)d_in[4];
    const float*      b1  = (const float*)d_in[5];
    const float*      W2  = (const float*)d_in[6];
    const float*      b2  = (const float*)d_in[7];
    const float*      W3  = (const float*)d_in[8];
    const float*      b3  = (const float*)d_in[9];
    const float*      W4  = (const float*)d_in[10];
    const float*      b4  = (const float*)d_in[11];

    // workspace layout (all 32B-aligned)
    float*    sums = (float*)d_ws;                                  // N*128 f32
    _Float16* W1p  = (_Float16*)((char*)d_ws + (size_t)NNODES * 128 * 4);
    _Float16* W2p  = W1p + 128 * 256;
    _Float16* W3p  = W2p + 256 * 256;
    _Float16* W4h  = W3p + 256 * 256;
    float*    acc  = (float*)(W4h + 256);

    hipMemsetAsync(sums, 0, (size_t)NNODES * 128 * sizeof(float), stream);
    hipMemsetAsync(acc, 0, sizeof(float), stream);

    k_cvt_weights<<<(256 * 256 + 255) / 256, 256, 0, stream>>>(
        W1, W2, W3, W4, W1p, W2p, W3p, W4h);

    k_scatter<<<(NEDGES * 32 + 255) / 256, 256, 0, stream>>>(x, pos, z, ei, sums);

    k_mlp<<<NNODES / 16, 128, 0, stream>>>(sums, W1p, b1, W2p, b2, W3p, b3,
                                           W4h, b4, acc);

    k_finalize<<<1, 1, 0, stream>>>(acc, (float*)d_out);
}
